// TriangularAttention_17016660426887
// MI455X (gfx1250) — matrix operand order repaired and checked by reading, not compiled, not run on hardware
//
#include <hip/hip_runtime.h>
#include <hip/hip_bf16.h>
#include <math.h>

typedef __attribute__((ext_vector_type(16))) _Float16 v16h;
typedef __attribute__((ext_vector_type(8)))  _Float16 v8h;
typedef __attribute__((ext_vector_type(8)))  float    v8f;

#define LSEQ 256
#define DM   128
#define NH   4
#define PAIRS ((size_t)LSEQ * LSEQ)

// Build a WMMA A/B fragment from two contiguous 8-half (16B) chunks.
__device__ __forceinline__ v16h ld16(const _Float16* lo, const _Float16* hi) {
    v8h a = *(const v8h*)lo;
    v8h b = *(const v8h*)hi;
    v16h r;
#pragma unroll
    for (int i = 0; i < 8; ++i) { r[i] = a[i]; r[i + 8] = b[i]; }
    return r;
}

__device__ __forceinline__ v8f wmma_f16(v16h a, v16h b, v8f c) {
    return __builtin_amdgcn_wmma_f32_16x16x32_f16(false, a, false, b, (short)0, c, false, false);
}

// ---------------------------------------------------------------------------
// Kernel 0: transpose + convert a 128x128 f32 weight to f16 column-major:
// Wt[c*128 + k] = W[k*128 + c].  B fragments then load as 2x16B contiguous.
// ---------------------------------------------------------------------------
__global__ __launch_bounds__(256)
void k0_wt(const float* __restrict__ W, _Float16* __restrict__ Wt)
{
    const int idx = blockIdx.x * 256 + threadIdx.x;   // 16384 elements
    const int k = idx >> 7, c = idx & 127;
    Wt[c * DM + k] = (_Float16)W[idx];
}

// ---------------------------------------------------------------------------
// Kernel 1: LayerNorm + projections q,k,v,gate (WMMA) + pair bias (VALU dots)
// Block = 128 threads (4 waves), 16 pairs per block.
// V and gate stored transposed ([dv_global][pair]) with packed 16B stores.
// ---------------------------------------------------------------------------
__global__ __launch_bounds__(128)
void k1_ln_proj(const float* __restrict__ z, const float* __restrict__ lnw,
                const float* __restrict__ lnb,
                const _Float16* __restrict__ wqt, const _Float16* __restrict__ wkt,
                const _Float16* __restrict__ wvt, const _Float16* __restrict__ wgt,
                const float* __restrict__ Wb, const float* __restrict__ bg,
                _Float16* __restrict__ qh, _Float16* __restrict__ kh,
                _Float16* __restrict__ vt, _Float16* __restrict__ gt,
                float* __restrict__ biasw)
{
    __shared__ float red[128];
    __shared__ float mean_s[16];
    __shared__ float rstd_s[16];
    __shared__ __align__(16) _Float16 zn[16 * DM];

    const int t  = threadIdx.x;
    const int p0 = blockIdx.x * 16;
    const int r  = t >> 3;
    const int c0 = (t & 7) * 16;
    const float* zrow = z + (size_t)(p0 + r) * DM;

    float s = 0.f;
#pragma unroll
    for (int u = 0; u < 16; ++u) s += zrow[c0 + u];
    red[t] = s;
    __syncthreads();
    if ((t & 7) == 0) {
        float m = 0.f;
#pragma unroll
        for (int u = 0; u < 8; ++u) m += red[(r << 3) + u];
        mean_s[r] = m * (1.f / 128.f);
    }
    __syncthreads();
    const float mu = mean_s[r];
    float vv = 0.f;
#pragma unroll
    for (int u = 0; u < 16; ++u) { float d = zrow[c0 + u] - mu; vv += d * d; }
    red[t] = vv;
    __syncthreads();
    if ((t & 7) == 0) {
        float m = 0.f;
#pragma unroll
        for (int u = 0; u < 8; ++u) m += red[(r << 3) + u];
        rstd_s[r] = rsqrtf(m * (1.f / 128.f) + 1e-5f);
    }
    __syncthreads();
    const float rstd = rstd_s[r];
#pragma unroll
    for (int u = 0; u < 16; ++u) {
        int c = c0 + u;
        zn[r * DM + c] = (_Float16)((zrow[c] - mu) * rstd * lnw[c] + lnb[c]);
    }
    __syncthreads();

    // pair bias head: 16 rows x 4 heads, 128-deep dots (threads 0..63)
    if (t < 64) {
        int rr = t >> 2, hh = t & 3;
        float acc = 0.f;
#pragma unroll 8
        for (int c = 0; c < 128; ++c) acc += (float)zn[rr * DM + c] * Wb[c * 4 + hh];
        biasw[(size_t)(p0 + rr) * 4 + hh] = acc;
    }

    // WMMA projections: wave w -> {q,k,v,gate}.  Force wave-uniform branching.
    const int wvu  = __builtin_amdgcn_readfirstlane(t >> 5);
    const int lane = t & 31;
    const int m    = lane & 15;
    const int hi   = lane >> 4;
    const _Float16* Wt = (wvu == 0) ? wqt : (wvu == 1) ? wkt : (wvu == 2) ? wvt : wgt;

    v16h aZ[4];
#pragma unroll
    for (int ks = 0; ks < 4; ++ks) {
        const _Float16* base = &zn[m * DM + ks * 32 + hi * 16];
        aZ[ks] = ld16(base, base + 8);
    }

#pragma unroll
    for (int nt = 0; nt < 8; ++nt) {
        const int col = nt * 16 + m;
        v8f c = {};
#pragma unroll
        for (int ks = 0; ks < 4; ++ks) {
            const _Float16* wb = Wt + (size_t)col * DM + ks * 32 + hi * 16;
            c = wmma_f16(aZ[ks], ld16(wb, wb + 8), c);
        }
        if (wvu == 0) {
#pragma unroll
            for (int e = 0; e < 8; ++e)
                qh[(size_t)(p0 + e + 8 * hi) * DM + col] = (_Float16)c[e];
        } else if (wvu == 1) {
#pragma unroll
            for (int e = 0; e < 8; ++e)
                kh[(size_t)(p0 + e + 8 * hi) * DM + col] = (_Float16)c[e];
        } else if (wvu == 2) {
            v8h pk;
#pragma unroll
            for (int e = 0; e < 8; ++e) pk[e] = (_Float16)c[e];
            *(v8h*)(vt + (size_t)col * PAIRS + p0 + 8 * hi) = pk;
        } else {
            const float bgv = bg[col];
            v8h pk;
#pragma unroll
            for (int e = 0; e < 8; ++e)
                pk[e] = (_Float16)__builtin_amdgcn_rcpf(1.f + __expf(-(c[e] + bgv)));
            *(v8h*)(gt + (size_t)col * PAIRS + p0 + 8 * hi) = pk;
        }
    }
}

// ---------------------------------------------------------------------------
// Kernel 2: row-wise attention, flash-style online softmax.
// Grid (r, h) = (256, 4); block = 512 threads = 16 waves; wave -> one 16-i tile.
// ---------------------------------------------------------------------------
__global__ __launch_bounds__(512)
void k2_attn(const _Float16* __restrict__ qh, const _Float16* __restrict__ kh,
             const _Float16* __restrict__ vt, const _Float16* __restrict__ gt,
             const float* __restrict__ biasw, const float* __restrict__ mask,
             _Float16* __restrict__ atth)
{
    __shared__ __align__(16) _Float16 pbuf[16][16 * 32];
    const int r    = blockIdx.x;
    const int h    = blockIdx.y;
    const int wv   = threadIdx.x >> 5;
    const int lane = threadIdx.x & 31;
    const int m    = lane & 15;
    const int hi   = lane >> 4;
    _Float16* pb   = &pbuf[wv][0];
    const float rscale = 0.17677669529663687f; // 1/sqrt(32)

    // Q A-fragment (16 i-rows x 32 dk)
    const size_t qoff = ((size_t)r * LSEQ + wv * 16 + m) * DM + h * 32;
    const v16h aQ = ld16(qh + qoff + hi * 16, qh + qoff + 8 + hi * 16);

    float mrun[8], lrun[8], maskI[8];
    v8f acc0 = {}, acc1 = {};
#pragma unroll
    for (int e = 0; e < 8; ++e) {
        mrun[e] = -1e30f; lrun[e] = 0.f;
        maskI[e] = mask[wv * 16 + e + 8 * hi];
    }

    const _Float16* v0base = vt + (size_t)(h * 32 + m) * PAIRS + (size_t)r * LSEQ;
    const _Float16* v1base = vt + (size_t)(h * 32 + 16 + m) * PAIRS + (size_t)r * LSEQ;

    for (int jt = 0; jt < 8; ++jt) {               // 32-wide j chunks
        if (jt < 7) {  // prefetch next chunk's K rows and V columns
            __builtin_prefetch(kh + ((size_t)r * LSEQ + (jt + 1) * 32 + m) * DM + h * 32, 0, 3);
            __builtin_prefetch(v0base + (jt + 1) * 32, 0, 3);
        }
        v8f c0, c1;
        {
            size_t koff = ((size_t)r * LSEQ + jt * 32 + m) * DM + h * 32 + hi * 16;
            v8f z0 = {};
            c0 = wmma_f16(aQ, ld16(kh + koff, kh + koff + 8), z0);
        }
        {
            size_t koff = ((size_t)r * LSEQ + jt * 32 + 16 + m) * DM + h * 32 + hi * 16;
            v8f z1 = {};
            c1 = wmma_f16(aQ, ld16(kh + koff, kh + koff + 8), z1);
        }
        const int j0 = jt * 32 + m, j1 = jt * 32 + 16 + m;
        const float mj0 = mask[j0], mj1 = mask[j1];
#pragma unroll
        for (int e = 0; e < 8; ++e) {
            const int il = e + 8 * hi;
            const size_t brow = (size_t)(wv * 16 + il) * LSEQ;
            float l0 = (maskI[e] * mj0 > 0.f) ? c0[e] * rscale + biasw[(brow + j0) * 4 + h] : -1e30f;
            float l1 = (maskI[e] * mj1 > 0.f) ? c1[e] * rscale + biasw[(brow + j1) * 4 + h] : -1e30f;
            float tmax = fmaxf(l0, l1);
            tmax = fmaxf(tmax, __shfl_xor(tmax, 1));
            tmax = fmaxf(tmax, __shfl_xor(tmax, 2));
            tmax = fmaxf(tmax, __shfl_xor(tmax, 4));
            tmax = fmaxf(tmax, __shfl_xor(tmax, 8));
            float mnew = fmaxf(mrun[e], tmax);
            float sc = __expf(mrun[e] - mnew);
            float p0 = (l0 > -5e29f) ? __expf(l0 - mnew) : 0.f;
            float p1 = (l1 > -5e29f) ? __expf(l1 - mnew) : 0.f;
            float rsum = p0 + p1;
            rsum += __shfl_xor(rsum, 1);
            rsum += __shfl_xor(rsum, 2);
            rsum += __shfl_xor(rsum, 4);
            rsum += __shfl_xor(rsum, 8);
            lrun[e] = lrun[e] * sc + rsum;
            mrun[e] = mnew;
            acc0[e] *= sc;
            acc1[e] *= sc;
            pb[il * 32 + m]      = (_Float16)p0;   // C-layout -> row-major via LDS
            pb[il * 32 + 16 + m] = (_Float16)p1;
        }
        asm volatile("s_wait_dscnt 0" ::: "memory"); // wave-private LDS RAW fence
        const v16h aP = ld16(pb + m * 32 + hi * 16, pb + m * 32 + 8 + hi * 16);
        // V B-fragments: transposed layout -> contiguous over j (kk = e + 16*hi)
        const _Float16* b0 = v0base + jt * 32 + 16 * hi;
        const _Float16* b1 = v1base + jt * 32 + 16 * hi;
        acc0 = wmma_f16(aP, ld16(b0, b0 + 8), acc0);
        acc1 = wmma_f16(aP, ld16(b1, b1 + 8), acc1);
    }

    // Gate: transposed layout -> one packed 16B load per output column
    const size_t pcol = (size_t)r * LSEQ + wv * 16 + 8 * hi;
    const v8h g0 = *(const v8h*)(gt + (size_t)(h * 32 + m) * PAIRS + pcol);
    const v8h g1 = *(const v8h*)(gt + (size_t)(h * 32 + 16 + m) * PAIRS + pcol);
#pragma unroll
    for (int e = 0; e < 8; ++e) {
        const int il = e + 8 * hi;
        const float inv = (lrun[e] > 0.f) ? __builtin_amdgcn_rcpf(lrun[e]) : 0.f;
        const size_t po = ((size_t)r * LSEQ + wv * 16 + il) * DM + h * 32;
        atth[po + m]      = (_Float16)(acc0[e] * inv * (float)g0[e]);
        atth[po + 16 + m] = (_Float16)(acc1[e] * inv * (float)g1[e]);
    }
}

// ---------------------------------------------------------------------------
// Kernel 3: output projection (gated flat) @ Wo + bo  -> f32 d_out
// Block = 128 threads (4 waves), 16 rows per block; wave w -> N-tiles 2w,2w+1
// ---------------------------------------------------------------------------
__global__ __launch_bounds__(128)
void k3_out(const _Float16* __restrict__ atth, const _Float16* __restrict__ wot,
            const float* __restrict__ bo, float* __restrict__ out)
{
    const int t    = threadIdx.x;
    const int p0   = blockIdx.x * 16;
    const int wv   = t >> 5;
    const int lane = t & 31;
    const int m    = lane & 15;
    const int hi   = lane >> 4;

    v16h aA[4];
#pragma unroll
    for (int ks = 0; ks < 4; ++ks) {
        const _Float16* base = atth + (size_t)(p0 + m) * DM + ks * 32 + hi * 16;
        aA[ks] = ld16(base, base + 8);
    }

#pragma unroll
    for (int q = 0; q < 2; ++q) {
        const int nt  = wv * 2 + q;
        const int col = nt * 16 + m;
        v8f c = {};
#pragma unroll
        for (int ks = 0; ks < 4; ++ks) {
            const _Float16* wb = wot + (size_t)col * DM + ks * 32 + hi * 16;
            c = wmma_f16(aA[ks], ld16(wb, wb + 8), c);
        }
        const float bb = bo[col];
#pragma unroll
        for (int e = 0; e < 8; ++e)
            out[(size_t)(p0 + e + 8 * hi) * DM + col] = c[e] + bb;
    }
}

// ---------------------------------------------------------------------------
extern "C" void kernel_launch(void* const* d_in, const int* in_sizes, int n_in,
                              void* d_out, int out_size, void* d_ws, size_t ws_size,
                              hipStream_t stream)
{
    (void)in_sizes; (void)n_in; (void)out_size; (void)ws_size;
    const float* z    = (const float*)d_in[0];
    const float* mask = (const float*)d_in[1];
    const float* lnw  = (const float*)d_in[2];
    const float* lnb  = (const float*)d_in[3];
    const float* Wq   = (const float*)d_in[4];
    const float* Wk   = (const float*)d_in[5];
    const float* Wv   = (const float*)d_in[6];
    const float* Wb   = (const float*)d_in[7];
    const float* Wg   = (const float*)d_in[8];
    const float* bg   = (const float*)d_in[9];
    const float* Wo   = (const float*)d_in[10];
    const float* bo   = (const float*)d_in[11];
    float* out = (float*)d_out;

    _Float16* qh    = (_Float16*)d_ws;
    _Float16* kh    = qh + PAIRS * DM;
    _Float16* vt    = kh + PAIRS * DM;            // transposed: [dv_global][pair]
    _Float16* gt    = vt + PAIRS * DM;            // transposed: [dv_global][pair]
    _Float16* atth  = gt + PAIRS * DM;
    float*    biasw = (float*)(atth + PAIRS * DM);
    _Float16* wqt   = (_Float16*)(biasw + PAIRS * 4);
    _Float16* wkt   = wqt + DM * DM;
    _Float16* wvt   = wkt + DM * DM;
    _Float16* wgt   = wvt + DM * DM;
    _Float16* wot   = wgt + DM * DM;              // total ws use: ~81.2 MB

    k0_wt<<<64, 256, 0, stream>>>(Wq, wqt);
    k0_wt<<<64, 256, 0, stream>>>(Wk, wkt);
    k0_wt<<<64, 256, 0, stream>>>(Wv, wvt);
    k0_wt<<<64, 256, 0, stream>>>(Wg, wgt);
    k0_wt<<<64, 256, 0, stream>>>(Wo, wot);

    k1_ln_proj<<<(unsigned)(PAIRS / 16), 128, 0, stream>>>(z, lnw, lnb, wqt, wkt, wvt,
                                                           wgt, Wb, bg, qh, kh, vt, gt, biasw);
    k2_attn<<<dim3(LSEQ, NH), 512, 0, stream>>>(qh, kh, vt, gt, biasw, mask, atth);
    k3_out<<<(unsigned)(PAIRS / 16), 128, 0, stream>>>(atth, wot, bo, out);
}
